// se_equivariant_net_75058848465296
// MI455X (gfx1250) — compile-verified, matrix operand order checked by reading, not run on hardware
//
#include <hip/hip_runtime.h>
#include <hip/hip_bf16.h>

#define NB 4
#define NN 256
#define ND 3
#define NH 64
#define NM 128

typedef __attribute__((ext_vector_type(16))) __bf16 v16bf;
typedef __attribute__((ext_vector_type(8)))  __bf16 v8bf;
typedef __attribute__((ext_vector_type(8)))  float  v8f;

__device__ __forceinline__ unsigned short f2bf(float f) {
  unsigned u = __float_as_uint(f);
  u += 0x7FFFu + ((u >> 16) & 1u);       // round-to-nearest-even
  return (unsigned short)(u >> 16);
}
__device__ __forceinline__ float bf2f(unsigned short h) {
  return __uint_as_float(((unsigned)h) << 16);
}

struct LayerParams {
  const float *we1, *be1, *we2, *be2, *winf, *binf;
  const float *wx1, *bx1, *wx2, *bx2, *wx3, *bx3;
  const float *wh1, *bh1, *wh2, *bh2, *wh3, *bh3;
};

// ---- dynamic LDS layout (ushort region then float region) ----
constexpr int HJ_OFF   = 0;                    // [256][64]  bf16 h_j
constexpr int ACTA_OFF = HJ_OFF + NN*NH;       // [256][128] bf16 activations A
constexpr int ACTB_OFF = ACTA_OFF + NN*NM;     // [256][128] bf16 activations B (m)
constexpr int WT_OFF   = ACTB_OFF + NN*NM;     // [128][K]   bf16 transposed weights
constexpr int US_TOTAL = WT_OFF + NM*NM;
constexpr int F_TOTAL  = NN*4 /*diffs*/ + NN /*sq*/ + NN /*invs*/ + NN /*e*/
                       + NM /*thi*/ + NM /*srow*/ + NM /*bias*/ + NM /*vec*/
                       + NM /*mi*/ + NH /*hif*/ + NM /*g1*/ + NM /*g2*/
                       + NN*4 /*red*/;
constexpr size_t SMEM_BYTES = (size_t)US_TOTAL*2 + (size_t)F_TOTAL*4;

// Generic tiled GEMM: Out[256][128] = epi(Ain[256][KDIM] @ WT[128][KDIM]^T)
// EPI 0: + sq[m]*srow[n] + thi[n] + bias[n], relu       (edge MLP layer 1)
// EPI 1: + bias[n], zero diagonal row m==i_node          (edge MLP layer 2 -> m)
// EPI 2: + bias[n], relu                                 (phi_x hidden layers)
template<int KDIM, int EPI>
__device__ __forceinline__ void gemm_block(
    const unsigned short* Ain, const unsigned short* WT, unsigned short* Out,
    const float* bias, const float* srow, const float* thi, const float* sqv,
    int i_node, int wave, int lane)
{
  const int half = lane >> 4;
  const int lx   = lane & 15;
  for (int mi = 0; mi < 2; ++mi) {
    const int mt  = wave * 2 + mi;            // 16-row j tile
    const int row = mt * 16 + lx;             // A row held by this lane
    const unsigned short* arow = Ain + row * KDIM;
    for (int nt = 0; nt < 8; ++nt) {
      const int n = nt * 16 + lx;             // output column held by this lane
      const unsigned short* brow = WT + n * KDIM;
      v8f acc = {};
#pragma unroll
      for (int ks = 0; ks < KDIM / 32; ++ks) {
        const int k0 = ks * 32 + half * 8;
        v8bf alo = *(const v8bf*)(arow + k0);
        v8bf ahi = *(const v8bf*)(arow + k0 + 16);
        v8bf blo = *(const v8bf*)(brow + k0);
        v8bf bhi = *(const v8bf*)(brow + k0 + 16);
        v16bf a  = __builtin_shufflevector(alo, ahi, 0,1,2,3,4,5,6,7,8,9,10,11,12,13,14,15);
        v16bf bm = __builtin_shufflevector(blo, bhi, 0,1,2,3,4,5,6,7,8,9,10,11,12,13,14,15);
        acc = __builtin_amdgcn_wmma_f32_16x16x32_bf16(
            /*neg_a=*/false, a, /*neg_b=*/false, bm,
            /*c_mod=*/(short)0, acc, /*reuse_a=*/false, /*reuse_b=*/false);
      }
#pragma unroll
      for (int r = 0; r < 8; ++r) {
        const int m = mt * 16 + r + half * 8;
        float v = acc[r];
        if (EPI == 0) v += sqv[m] * srow[n] + thi[n] + bias[n];
        else          v += bias[n];
        if (EPI == 0 || EPI == 2) v = v > 0.f ? v : 0.f;
        if (EPI == 1 && m == i_node) v = 0.f;
        Out[m * NM + n] = f2bf(v);
      }
    }
  }
}

__device__ __forceinline__ void stage_wt128(const float* W, unsigned short* wt, int tid) {
  // W row-major [128][128] -> wt transposed [f][k] (bf16)
  for (int idx = tid; idx < NM * NM; idx += 256) {
    const int k = idx >> 7, f = idx & 127;
    wt[f * NM + k] = f2bf(W[idx]);
  }
}

__global__ void egnn_init_kernel(const float* __restrict__ x,
                                 const float* __restrict__ ew,
                                 const float* __restrict__ eb,
                                 float* __restrict__ xbuf,
                                 float* __restrict__ hbuf)
{
  const int b = blockIdx.x, i = threadIdx.x;
  const float* xb = x + b * NN * ND;
  const float xi0 = xb[i*3+0], xi1 = xb[i*3+1], xi2 = xb[i*3+2];
  float s = 0.f;
  for (int j = 0; j < NN; ++j) {
    float d0 = xb[j*3+0]-xi0, d1 = xb[j*3+1]-xi1, d2 = xb[j*3+2]-xi2;
    if (j != i) s += d0*d0 + d1*d1 + d2*d2;
  }
  const float ms = s * (1.f / NN);
  for (int f = 0; f < NH; ++f)
    hbuf[(b*NN + i)*NH + f] = ms * ew[f] + eb[f];
  xbuf[(b*NN+i)*3+0] = xi0; xbuf[(b*NN+i)*3+1] = xi1; xbuf[(b*NN+i)*3+2] = xi2;
}

__global__ void __launch_bounds__(256, 1)
egnn_layer_kernel(const float* __restrict__ x_in, const float* __restrict__ h_in,
                  float* __restrict__ x_out, float* __restrict__ h_out,
                  LayerParams p)
{
  extern __shared__ __align__(16) char smem_raw[];
  unsigned short* usS  = (unsigned short*)smem_raw;
  unsigned short* hjS  = usS + HJ_OFF;
  unsigned short* actA = usS + ACTA_OFF;
  unsigned short* actB = usS + ACTB_OFF;
  unsigned short* wtS  = usS + WT_OFF;
  float* fS    = (float*)(usS + US_TOTAL);
  float* diffs = fS;                 // [256][4]
  float* sqv   = diffs + NN*4;       // [256]
  float* invs  = sqv + NN;           // [256] 1/(sq+1)
  float* ev    = invs + NN;          // [256] gate
  float* thiS  = ev + NN;            // [128] h_i @ we1_hi
  float* srowS = thiS + NM;          // [128] we1 row 0 (sq row)
  float* biasS = srowS + NM;         // [128] current bias
  float* vecS  = biasS + NM;         // [128] winf then wx3
  float* miS   = vecS + NM;          // [128] m_i
  float* hifS  = miS + NM;           // [64]  h_i (f32)
  float* g1S   = hifS + NH;          // [128]
  float* g2S   = g1S + NM;           // [128]
  float* redS  = g2S + NM;           // [256][4]

  const int tid  = threadIdx.x;
  const int wave = tid >> 5, lane = tid & 31;
  const int b = blockIdx.x >> 8, i = blockIdx.x & 255;
  const float* xb = x_in + b * NN * ND;
  const float* hb = h_in + b * NN * NH;

  __builtin_prefetch(p.we2, 0, 0);
  __builtin_prefetch(p.wx1, 0, 0);

  // ---- Phase A: stage h_j (bf16), h_i, diff/sq; WT1 (we1 rows 1..64 transposed)
  for (int idx = tid; idx < NN*NH; idx += 256) hjS[idx] = f2bf(hb[idx]);
  if (tid < NH) hifS[tid] = hb[i*NH + tid];
  {
    const int j = tid;
    const float xi0 = xb[i*3+0], xi1 = xb[i*3+1], xi2 = xb[i*3+2];
    float d0 = xb[j*3+0]-xi0, d1 = xb[j*3+1]-xi1, d2 = xb[j*3+2]-xi2;
    if (j == i) { d0 = d1 = d2 = 0.f; }
    const float s = d0*d0 + d1*d1 + d2*d2;
    diffs[j*4+0] = d0; diffs[j*4+1] = d1; diffs[j*4+2] = d2;
    sqv[j] = s; invs[j] = 1.f / (s + 1.f);
  }
  for (int idx = tid; idx < 64*NM; idx += 256) {
    const int k = idx >> 7, f = idx & 127;
    wtS[f*64 + k] = f2bf(p.we1[(1 + k)*NM + f]);
  }
  if (tid < NM) { srowS[tid] = p.we1[tid]; biasS[tid] = p.be1[tid]; }
  __syncthreads();

  // t_hi[f] = sum_k h_i[k] * we1[65+k][f]  (constant over j)
  if (tid < NM) {
    float acc = 0.f;
    for (int k = 0; k < NH; ++k) acc += hifS[k] * p.we1[(65 + k)*NM + tid];
    thiS[tid] = acc;
  }
  __syncthreads();

  // ---- GEMM1: a1 = relu(h_j@W + sq*srow + thi + b1) -> actA
  gemm_block<NH, 0>(hjS, wtS, actA, biasS, srowS, thiS, sqv, i, wave, lane);
  __syncthreads();

  // ---- stage we2^T, be2, winf
  stage_wt128(p.we2, wtS, tid);
  if (tid < NM) { biasS[tid] = p.be2[tid]; vecS[tid] = p.winf[tid]; }
  __syncthreads();

  // ---- GEMM2: m = a1@we2 + b2, diagonal zeroed -> actB
  gemm_block<NM, 1>(actA, wtS, actB, biasS, srowS, thiS, sqv, i, wave, lane);
  __syncthreads();

  // ---- gate e[j] = sigmoid(m . winf + binf), masked
  {
    const int j = tid;
    float acc = p.binf[0];
    for (int f = 0; f < NM; ++f) acc += bf2f(actB[j*NM + f]) * vecS[f];
    const float e = 1.f / (1.f + __expf(-acc));
    ev[j] = (j == i) ? 0.f : e;
  }
  __syncthreads();

  // ---- m_i[f] = sum_j m[j][f]*e[j]; stage wx1^T + bx1
  if (tid < NM) {
    float acc = 0.f;
    for (int j = 0; j < NN; ++j) acc += bf2f(actB[j*NM + tid]) * ev[j];
    miS[tid] = acc;
  }
  stage_wt128(p.wx1, wtS, tid);
  if (tid < NM) biasS[tid] = p.bx1[tid];
  __syncthreads();

  // ---- GEMM3: a2 = relu(m@wx1 + bx1) -> actA
  gemm_block<NM, 2>(actB, wtS, actA, biasS, srowS, thiS, sqv, i, wave, lane);
  __syncthreads();

  // ---- stage wx2^T + bx2, wx3
  stage_wt128(p.wx2, wtS, tid);
  if (tid < NM) { biasS[tid] = p.bx2[tid]; vecS[tid] = p.wx3[tid]; }
  __syncthreads();

  // ---- GEMM4: a3 = relu(a2@wx2 + bx2) -> actB
  gemm_block<NM, 2>(actA, wtS, actB, biasS, srowS, thiS, sqv, i, wave, lane);
  __syncthreads();

  // ---- px[j] and coordinate-shift partials
  {
    const int j = tid;
    float acc = p.bx3[0];
    for (int f = 0; f < NM; ++f) acc += bf2f(actB[j*NM + f]) * vecS[f];
    const float px = (j == i) ? 0.f : acc;
    const float w  = invs[j] * px;
    redS[j*4+0] = diffs[j*4+0] * w;
    redS[j*4+1] = diffs[j*4+1] * w;
    redS[j*4+2] = diffs[j*4+2] * w;
  }
  __syncthreads();
  for (int s = 128; s >= 1; s >>= 1) {
    if (tid < s) {
      redS[tid*4+0] += redS[(tid+s)*4+0];
      redS[tid*4+1] += redS[(tid+s)*4+1];
      redS[tid*4+2] += redS[(tid+s)*4+2];
    }
    __syncthreads();
  }
  if (tid < 3)
    x_out[(b*NN + i)*3 + tid] = xb[i*3 + tid] + redS[tid] * (1.f / (NN - 1));

  // ---- phi_h (per-node, scalar): h += ((relu([m_i,h_i]@wh1+b)@wh2+b)@wh3+b)
  if (tid < NM) {
    float acc = p.bh1[tid];
    for (int k = 0; k < NM; ++k) acc += miS[k]  * p.wh1[k*NM + tid];
    for (int k = 0; k < NH; ++k) acc += hifS[k] * p.wh1[(NM + k)*NM + tid];
    g1S[tid] = acc > 0.f ? acc : 0.f;
  }
  __syncthreads();
  if (tid < NM) {
    float acc = p.bh2[tid];
    for (int k = 0; k < NM; ++k) acc += g1S[k] * p.wh2[k*NM + tid];
    g2S[tid] = acc;
  }
  __syncthreads();
  if (tid < NH) {
    float acc = p.bh3[tid];
    for (int k = 0; k < NM; ++k) acc += g2S[k] * p.wh3[k*NH + tid];
    h_out[(b*NN + i)*NH + tid] = hifS[tid] + acc;
  }
}

extern "C" void kernel_launch(void* const* d_in, const int* in_sizes, int n_in,
                              void* d_out, int out_size, void* d_ws, size_t ws_size,
                              hipStream_t stream) {
  (void)in_sizes; (void)n_in; (void)out_size; (void)ws_size;
  const float* x  = (const float*)d_in[0];
  const float* ew = (const float*)d_in[1];
  const float* eb = (const float*)d_in[2];

  float* ws = (float*)d_ws;
  float* x0 = ws;
  float* x1 = x0 + NB*NN*ND;
  float* h0 = x1 + NB*NN*ND;
  float* h1 = h0 + NB*NN*NH;

  hipFuncSetAttribute((const void*)egnn_layer_kernel,
                      hipFuncAttributeMaxDynamicSharedMemorySize,
                      (int)SMEM_BYTES);

  egnn_init_kernel<<<NB, NN, 0, stream>>>(x, ew, eb, x0, h0);

  LayerParams P[3];
  for (int l = 0; l < 3; ++l) {
    const float* const* q = (const float* const*)(d_in + 3 + l*18);
    P[l].we1 = q[0];  P[l].be1 = q[1];  P[l].we2 = q[2];  P[l].be2 = q[3];
    P[l].winf = q[4]; P[l].binf = q[5]; P[l].wx1 = q[6];  P[l].bx1 = q[7];
    P[l].wx2 = q[8];  P[l].bx2 = q[9];  P[l].wx3 = q[10]; P[l].bx3 = q[11];
    P[l].wh1 = q[12]; P[l].bh1 = q[13]; P[l].wh2 = q[14]; P[l].bh2 = q[15];
    P[l].wh3 = q[16]; P[l].bh3 = q[17];
  }

  egnn_layer_kernel<<<NB*NN, 256, SMEM_BYTES, stream>>>(x0, h0, x1, h1, P[0]);
  egnn_layer_kernel<<<NB*NN, 256, SMEM_BYTES, stream>>>(x1, h1, x0, h0, P[1]);
  egnn_layer_kernel<<<NB*NN, 256, SMEM_BYTES, stream>>>(x0, h0, (float*)d_out, h1, P[2]);
}
